// PiT_79748952752793
// MI455X (gfx1250) — compile-verified, matrix operand order checked by reading, not run on hardware
//
#include <hip/hip_runtime.h>
#include <hip/hip_bf16.h>
#include <math.h>

typedef __attribute__((ext_vector_type(16))) _Float16 v16h;
typedef __attribute__((ext_vector_type(8)))  _Float16 v8h;
typedef __attribute__((ext_vector_type(8)))  float    v8f;
typedef __attribute__((ext_vector_type(4)))  unsigned u32x4;
typedef __attribute__((ext_vector_type(4)))  int      i32x4;
typedef __attribute__((ext_vector_type(8)))  int      i32x8;

#define HID 256
#define VD  32

// ROCm 7.2 (clang-22) exposes the 5-arg tensor_load_to_lds builtin (probe-verified).
#if defined(__clang_major__) && (__clang_major__ < 23) && \
    __has_builtin(__builtin_amdgcn_tensor_load_to_lds)
#define USE_TDM 1
#else
#define USE_TDM 0
#endif

__device__ __forceinline__ float gelu_tanh(float x) {
  const float k0 = 0.7978845608028654f; // sqrt(2/pi)
  float x3 = x * x * x;
  return 0.5f * x * (1.f + tanhf(k0 * (x + 0.044715f * x3)));
}

__device__ __forceinline__ unsigned pack_h2(float a, float b) {
  union { _Float16 h[2]; unsigned u; } p;
  p.h[0] = (_Float16)a;
  p.h[1] = (_Float16)b;
  return p.u;
}

// A-fragment slot for element (q, 2*kp) of a 16x32 f16 tile (packed pair).
__device__ __forceinline__ void afrag_slot(int q, int kp, int& lane, int& dw) {
  lane = q + (((kp & 7) >= 4) ? 16 : 0);
  dw = (kp & 3) + ((kp >= 8) ? 4 : 0);
}

// Assemble B fragment from a k-contiguous [col][k<=32] LDS row: two b128 loads.
__device__ __forceinline__ v16h bfrag_load(const _Float16* vb, int lane) {
  int kh = (lane >> 4) * 8;
  v8h blo = *(const v8h*)(vb + kh);
  v8h bhi = *(const v8h*)(vb + 16 + kh);
  v16h bf;
#pragma unroll
  for (int j = 0; j < 8; ++j) { bf[j] = blo[j]; bf[8 + j] = bhi[j]; }
  return bf;
}

#if USE_TDM
// TDM: 3D tile (x = 32 contiguous k, y = 32 cols stride Nk, z = 4 batches stride 256*Nk)
__device__ __forceinline__ void pa_tdm_issue(const _Float16* V, int h, int Nk, int k0,
                                             unsigned lds_off) {
  unsigned long long ga =
      (unsigned long long)(const void*)V + 2ull * ((size_t)(h * VD) * Nk + k0);
  unsigned dim0 = (unsigned)Nk, dim1 = 1024u;
  unsigned long long st0 = (unsigned long long)Nk;
  unsigned long long st1 = 256ull * (unsigned long long)Nk;
  u32x4 g0;
  g0[0] = 1u;                                   // count=1, user mode
  g0[1] = lds_off;                              // lds_addr
  g0[2] = (unsigned)(ga & 0xFFFFFFFFull);       // global_addr[31:0]
  g0[3] = (unsigned)((ga >> 32) & 0x1FFFFFFull) | (2u << 30);  // addr[56:32], type=2
  i32x8 g1;
  g1[0] = (int)(1u << 16);                      // data_size = 2B
  g1[1] = (int)((dim0 & 0xFFFFu) << 16);        // tensor_dim0[15:0]
  g1[2] = (int)((dim0 >> 16) | ((dim1 & 0xFFFFu) << 16));
  g1[3] = (int)((dim1 >> 16) | (32u << 16));    // tile_dim0 = 32
  g1[4] = (int)(32u | (4u << 16));              // tile_dim1 = 32, tile_dim2 = 4
  g1[5] = (int)(unsigned)(st0 & 0xFFFFFFFFull); // tensor_dim0_stride
  g1[6] = (int)((unsigned)(st0 >> 32) | ((unsigned)(st1 & 0xFFFFull) << 16));
  g1[7] = (int)(unsigned)(st1 >> 16);           // tensor_dim1_stride[47:16]
  i32x4 g2;
  g2[0] = 4; g2[1] = 0; g2[2] = 0; g2[3] = 0;   // tensor_dim2 = 4
  i32x4 g3;
  g3[0] = 0; g3[1] = 0; g3[2] = 0; g3[3] = 0;
  __builtin_amdgcn_tensor_load_to_lds(g0, g1, g2, g3, 0);
}
#endif

// ---------------- encoder: h = gelu(x @ en_w.T + en_b), K = 3 ----------------
__global__ void encoder_kernel(const float* __restrict__ x, const float* __restrict__ w,
                               const float* __restrict__ bvec, float* __restrict__ out,
                               int total) {
  int idx = blockIdx.x * blockDim.x + threadIdx.x;
  if (idx >= total) return;
  int c = idx & (HID - 1);
  int bn = idx >> 8;
  float a = bvec[c];
  a += x[bn * 3 + 0] * w[c * 3 + 0];
  a += x[bn * 3 + 1] * w[c * 3 + 1];
  a += x[bn * 3 + 2] * w[c * 3 + 2];
  out[idx] = gelu_tanh(a);
}

// ---------------- generic WMMA GEMM: [M,256] x [256,256], double-buffered ----------------
// bmode 0: B = head-concat value weights [8,256,32] -> B[k][c] = W[c/32][k][c%32]
// bmode 1: B = eqx linear weight [out,in]           -> B[k][c] = W[c][k]
// f16 output is written TRANSPOSED: outh[(b*256 + col)*nper + nrow].
__global__ __launch_bounds__(256) void wmma_gemm_kernel(
    const float* __restrict__ A, const float* __restrict__ B,
    const float* __restrict__ bias, const float* __restrict__ addend,
    float* __restrict__ outf, _Float16* __restrict__ outh,
    int bmode, int act, int nper) {
  __shared__ __align__(32) unsigned s_afrag[2][32][8];   // A in WMMA fragment layout
  __shared__ __align__(16) _Float16 sBT[2][128][32];     // B transposed: [col][k]
  int tid = threadIdx.x;
  int lane = tid & 31, w = tid >> 5;
  int m0 = blockIdx.x * 16;
  int c0 = blockIdx.y * 128;
  v8f acc = {0.f, 0.f, 0.f, 0.f, 0.f, 0.f, 0.f, 0.f};

  auto stage = [&](int buf, int k0) {
    {  // A tile 16x32 -> fragment layout, one packed dword store per thread
      int q = tid >> 4, kp = tid & 15;
      const float* ap = A + (size_t)(m0 + q) * HID + k0 + 2 * kp;
      float a0 = ap[0], a1 = ap[1];
      int ln, dw;
      afrag_slot(q, kp, ln, dw);
      s_afrag[buf][ln][dw] = pack_h2(a0, a1);
    }
    for (int i = tid; i < 2048; i += 256) {  // B tile 32x128 -> sBT[col][k]
      int c = i >> 4, kp = i & 15, k = 2 * kp;
      int col = c0 + c;
      float b0, b1;
      if (bmode == 0) {
        const float* bp =
            B + (size_t)(col >> 5) * (HID * VD) + (size_t)(k0 + k) * VD + (col & 31);
        b0 = bp[0];
        b1 = bp[VD];
      } else {
        const float* bp = B + (size_t)col * HID + (k0 + k);
        b0 = bp[0];
        b1 = bp[1];
      }
      ((unsigned*)&sBT[buf][c][0])[kp] = pack_h2(b0, b1);
    }
  };

  stage(0, 0);
  __syncthreads();
  const int T = HID / 32;
#pragma unroll
  for (int t = 0; t < T; ++t) {
    int cur = t & 1;
    if (t + 1 < T) stage(cur ^ 1, (t + 1) * 32);
    v16h af = *(const v16h*)&s_afrag[cur][lane][0];
    v16h bf = bfrag_load(&sBT[cur][w * 16 + (lane & 15)][0], lane);
    acc = __builtin_amdgcn_wmma_f32_16x16x32_f16(false, af, false, bf, (short)0, acc,
                                                 false, false);
    __syncthreads();
  }
  int n = lane & 15;
  int col = c0 + w * 16 + n;
#pragma unroll
  for (int i = 0; i < 8; ++i) {
    int m = i + 8 * (lane >> 4);
    int row = m0 + m;
    float v = acc[i];
    if (bias)   v += bias[col];
    if (addend) v += addend[(size_t)row * HID + col];
    if (act)    v = gelu_tanh(v);
    if (outh) {
      int b = row / nper, nr = row - b * nper;
      outh[((size_t)b * HID + col) * nper + nr] = (_Float16)v;  // transposed f16
    } else {
      outf[(size_t)row * HID + col] = v;
    }
  }
}

// ---------------- fused position-attention (double-buffered TDM) ----------------
// V layout: [4, 256, Nk] f16 (k-contiguous). Per workgroup: 16 q-rows, one head.
// Phase A: exact rank-r threshold via 4x8-bit radix select + row min.
// Phase B: att tile built in WMMA A-fragment layout; V tiles DMA'd to ping-pong LDS
// buffers by the Tensor Data Mover while the previous tile is consumed by WMMA.
__global__ __launch_bounds__(256) void posatt_kernel(
    const float* __restrict__ mdist,  // [8, Nq, Nk]
    const float* __restrict__ rvec,   // [8]
    const _Float16* __restrict__ V,   // [4, 256, Nk]
    float* __restrict__ out,          // [4, Nq, 256]
    int Nq, int Nk, int rank) {
  __shared__ unsigned s_hist[256];
  __shared__ unsigned s_scan[256];
  __shared__ unsigned s_sel2[2];
  __shared__ unsigned s_minb;
  __shared__ unsigned s_thr[16];
  __shared__ float    s_mmin[16];
  __shared__ float    s_rowsum[16];
  __shared__ __align__(32) unsigned s_afrag[2][32][8];   // att tile, fragment layout
  __shared__ __align__(16) _Float16 s_vT[2][4][32][32];  // [buf][b][col][k]

  int tid = threadIdx.x;
  int lane = tid & 31, w = tid >> 5;
  int q0 = blockIdx.x * 16;
  int h = blockIdx.y;
  float r = rvec[h];
  float scale = tanf(0.25f * 3.14159265358979f * (1.f - 1e-7f) * (1.f + sinf(r)));
  const float* md = mdist + ((size_t)h * Nq + q0) * Nk;

  if (tid < 16) s_rowsum[tid] = 0.f;

  // ---- Phase A: rank-r threshold + row min per q-row ----
  for (int q = 0; q < 16; ++q) {
    const float* row = md + (size_t)q * Nk;
    if (tid == 0) s_minb = 0xFFFFFFFFu;
    s_hist[tid] = 0u;
    __syncthreads();
    unsigned lmin = 0xFFFFFFFFu;
    for (int i = tid; i < Nk; i += 256) {
      unsigned v = __float_as_uint(row[i]);
      lmin = (v < lmin) ? v : lmin;
      atomicAdd(&s_hist[v >> 24], 1u);
    }
    atomicMin(&s_minb, lmin);
    __syncthreads();
    unsigned prefix = 0u;
    unsigned rrem = (unsigned)rank;
    for (int p = 0;; ++p) {
      unsigned c = s_hist[tid];
      s_scan[tid] = c;
      __syncthreads();
      for (int off = 1; off < 256; off <<= 1) {
        unsigned vv = (tid >= off) ? s_scan[tid - off] : 0u;
        __syncthreads();
        s_scan[tid] += vv;
        __syncthreads();
      }
      unsigned incl = s_scan[tid];
      unsigned excl = incl - c;
      if (c > 0u && rrem >= excl && rrem < incl) {
        s_sel2[0] = (unsigned)tid;
        s_sel2[1] = rrem - excl;
      }
      __syncthreads();
      prefix = (prefix << 8) | s_sel2[0];
      rrem = s_sel2[1];
      if (p == 3) break;
      int shift = 24 - 8 * (p + 1);
      s_hist[tid] = 0u;
      __syncthreads();
      for (int i = tid; i < Nk; i += 256) {
        unsigned v = __float_as_uint(row[i]);
        if ((v >> (shift + 8)) == prefix) atomicAdd(&s_hist[(v >> shift) & 255u], 1u);
      }
      __syncthreads();
    }
    if (tid == 0) {
      s_thr[q] = prefix;
      s_mmin[q] = __uint_as_float(s_minb);
    }
    __syncthreads();
  }

  // ---- Phase B ----
  int b = w >> 1;
  int nh = (w & 1) * 16;
  v8f acc = {0.f, 0.f, 0.f, 0.f, 0.f, 0.f, 0.f, 0.f};

  auto stage_att = [&](int buf, int k0) {
    int q = tid >> 4, kp = tid & 15, k = 2 * kp;
    const float* mp = md + (size_t)q * Nk + k0 + k;
    float m0v = mp[0], m1v = mp[1];
    if (k0 + 32 < Nk) __builtin_prefetch(mp + 32);  // global_prefetch next tile
    float p0 = 0.f, p1 = 0.f;
    if (__float_as_uint(m0v) <= s_thr[q]) p0 = __expf(-scale * (m0v - s_mmin[q]));
    if (__float_as_uint(m1v) <= s_thr[q]) p1 = __expf(-scale * (m1v - s_mmin[q]));
    int ln, dw;
    afrag_slot(q, kp, ln, dw);
    s_afrag[buf][ln][dw] = pack_h2(p0, p1);
    float ps = p0 + p1;
    if (ps != 0.f) atomicAdd(&s_rowsum[q], ps);
  };
  auto stage_v = [&](int buf, int k0) {
#if USE_TDM
    if (w == 0) pa_tdm_issue(V, h, Nk, k0, (unsigned)(size_t)&s_vT[buf][0][0][0]);
#else
    for (int i = tid; i < 512; i += 256) {
      int b2 = i >> 7, rem = i & 127;
      int n = rem >> 2, kk = rem & 3;
      *(uint4*)&s_vT[buf][b2][n][kk * 8] =
          *(const uint4*)&V[((size_t)(b2 * HID + h * VD + n)) * Nk + k0 + kk * 8];
    }
#endif
  };

  const int T = Nk >> 5;
  // prologue: tile 0 into buffer 0
  stage_v(0, 0);
  stage_att(0, 0);
#if USE_TDM
  if (w == 0) __builtin_amdgcn_s_wait_tensorcnt(0);
#endif
  __syncthreads();
  for (int t = 0; t < T; ++t) {
    int cur = t & 1;
    if (t + 1 < T) {
      stage_v(cur ^ 1, (t + 1) * 32);   // DMA next tile while computing this one
      stage_att(cur ^ 1, (t + 1) * 32);
    }
    v16h af = *(const v16h*)&s_afrag[cur][lane][0];
    v16h bf = bfrag_load(&s_vT[cur][b][nh + (lane & 15)][0], lane);
    acc = __builtin_amdgcn_wmma_f32_16x16x32_f16(false, af, false, bf, (short)0, acc,
                                                 false, false);
#if USE_TDM
    if (w == 0 && t + 1 < T) __builtin_amdgcn_s_wait_tensorcnt(0);
#endif
    __syncthreads();
  }
  int n = lane & 15;
#pragma unroll
  for (int i = 0; i < 8; ++i) {
    int m = i + 8 * (lane >> 4);
    float v = acc[i] / s_rowsum[m];  // deferred softmax normalization
    v = gelu_tanh(v);
    out[((size_t)b * Nq + q0 + m) * HID + h * VD + nh + n] = v;
  }
}

// ---------------- final projection to OUT_CH = 1 ----------------
__global__ void head_proj_kernel(const float* __restrict__ hbuf, const float* __restrict__ w,
                                 const float* __restrict__ bvec, float* __restrict__ out,
                                 int rows) {
  int rr = blockIdx.x * blockDim.x + threadIdx.x;
  if (rr >= rows) return;
  float a = bvec[0];
  const float* hp = hbuf + (size_t)rr * HID;
#pragma unroll 4
  for (int c = 0; c < HID; ++c) a += hp[c] * w[c];
  out[rr] = a;
}

extern "C" void kernel_launch(void* const* d_in, const int* in_sizes, int n_in,
                              void* d_out, int out_size, void* d_ws, size_t ws_size,
                              hipStream_t stream) {
  (void)in_sizes; (void)n_in; (void)out_size; (void)ws_size;
  const float* x      = (const float*)d_in[0];
  const float* mdd    = (const float*)d_in[1];
  const float* mdb    = (const float*)d_in[2];
  const float* mdu    = (const float*)d_in[3];
  const float* en_w   = (const float*)d_in[4];
  const float* en_b   = (const float*)d_in[5];
  const float* down_r = (const float*)d_in[6];
  const float* down_w = (const float*)d_in[7];
  const float* pa_r   = (const float*)d_in[8];
  const float* pa_w   = (const float*)d_in[9];
  const float* mlp1_w = (const float*)d_in[10];
  const float* mlp1_b = (const float*)d_in[11];
  const float* mlp2_w = (const float*)d_in[12];
  const float* mlp2_b = (const float*)d_in[13];
  const float* res_w  = (const float*)d_in[14];
  const float* res_b  = (const float*)d_in[15];
  const float* up_r   = (const float*)d_in[16];
  const float* up_w   = (const float*)d_in[17];
  const float* de1_w  = (const float*)d_in[18];
  const float* de1_b  = (const float*)d_in[19];
  const float* de2_w  = (const float*)d_in[20];
  const float* de2_b  = (const float*)d_in[21];

  const int B = 4, NF = 4096, NL = 1024;
  float* h_full  = (float*)d_ws;                        // [B,NF,HID] f32
  float* h_full2 = h_full  + (size_t)B * NF * HID;      // [B,NF,HID] f32
  float* L0      = h_full2 + (size_t)B * NF * HID;      // [B,NL,HID] f32
  float* L1      = L0 + (size_t)B * NL * HID;
  float* t1      = L1 + (size_t)B * NL * HID;
  float* t2      = t1 + (size_t)B * NL * HID;
  _Float16* Vh   = (_Float16*)(t2 + (size_t)B * NL * HID);  // [B,256,max Nk] f16

  // 1) encoder lift
  {
    int total = B * NF * HID;
    encoder_kernel<<<(total + 255) / 256, 256, 0, stream>>>(x, en_w, en_b, h_full, total);
  }
  // 2) value projection for down-PA (f16, transposed [b][col][k])
  wmma_gemm_kernel<<<dim3(B * NF / 16, 2), 256, 0, stream>>>(
      h_full, down_w, nullptr, nullptr, nullptr, Vh, 0, 0, NF);
  // 3) down PA: Nq=1024, Nk=4096, rank = floor(0.03*4095) = 122
  posatt_kernel<<<dim3(NL / 16, 8), 256, 0, stream>>>(mdd, down_r, Vh, L0, NL, NF, 122);

  float* cur = L0;
  float* alt = L1;
  for (int i = 0; i < 4; ++i) {
    wmma_gemm_kernel<<<dim3(B * NL / 16, 2), 256, 0, stream>>>(
        cur, pa_w + (size_t)i * 8 * HID * VD, nullptr, nullptr, nullptr, Vh, 0, 0, NL);
    // rank = floor(0.05*1023) = 51
    posatt_kernel<<<dim3(NL / 16, 8), 256, 0, stream>>>(
        mdb + (size_t)i * 8 * NL * NL, pa_r + (size_t)i * 8, Vh, alt, NL, NL, 51);
    wmma_gemm_kernel<<<dim3(B * NL / 16, 2), 256, 0, stream>>>(
        alt, mlp1_w + (size_t)i * HID * HID, mlp1_b + (size_t)i * HID, nullptr,
        t1, nullptr, 1, 1, NL);
    wmma_gemm_kernel<<<dim3(B * NL / 16, 2), 256, 0, stream>>>(
        t1, mlp2_w + (size_t)i * HID * HID, mlp2_b + (size_t)i * HID, nullptr,
        t2, nullptr, 1, 0, NL);
    // h = gelu(res(h) + mlp_out)
    wmma_gemm_kernel<<<dim3(B * NL / 16, 2), 256, 0, stream>>>(
        cur, res_w + (size_t)i * HID * HID, res_b + (size_t)i * HID, t2,
        alt, nullptr, 1, 1, NL);
    float* tmp = cur; cur = alt; alt = tmp;
  }

  // up-PA value projection
  wmma_gemm_kernel<<<dim3(B * NL / 16, 2), 256, 0, stream>>>(
      cur, up_w, nullptr, nullptr, nullptr, Vh, 0, 0, NL);
  // up PA: Nq=4096, Nk=1024, rank = floor(0.03*1023) = 30
  posatt_kernel<<<dim3(NF / 16, 8), 256, 0, stream>>>(mdu, up_r, Vh, h_full, NF, NL, 30);
  // decoder linear + gelu
  wmma_gemm_kernel<<<dim3(B * NF / 16, 2), 256, 0, stream>>>(
      h_full, de1_w, de1_b, nullptr, h_full2, nullptr, 1, 1, NF);
  // final projection to 1 channel
  head_proj_kernel<<<(B * NF + 255) / 256, 256, 0, stream>>>(
      h_full2, de2_w, de2_b, (float*)d_out, B * NF);
}
